// AttentionAggregation_78219944395235
// MI455X (gfx1250) — compile-verified, matrix operand order checked by reading, not run on hardware
//
#include <hip/hip_runtime.h>

// ---------------------------------------------------------------------------
// AttentionAggregation for MI455X (gfx1250, wave32, WMMA 16x16x32 f16->f32)
//
// B=4, N=4096, D=128.  ~36 GFLOP of GEMM, ~8MB of data -> compute bound ONLY
// if K/V are not privately re-streamed per wave.  Block-shared K/V tiles in
// LDS, double-buffered via GLOBAL_LOAD_ASYNC_TO_LDS (ASYNCcnt fenced),
// cutting L2 traffic 8x.  All matmuls on v_wmma_f32_16x16x32_f16.
// ---------------------------------------------------------------------------

typedef __attribute__((ext_vector_type(16))) _Float16 v16h;
typedef __attribute__((ext_vector_type(8)))  _Float16 v8h;
typedef __attribute__((ext_vector_type(8)))  float    v8f;
typedef __attribute__((ext_vector_type(4)))  int      i32x4;

#define DHEAD 128
#define NSEQ  4096
#define NB    4
#define QTPB  (NSEQ / 16)   // 256 query tiles per batch
#define KC    32            // keys staged per chunk
#define NCHUNK (NSEQ / KC)  // 128

// ---- CDNA5 async global->LDS copy (guarded; sync fallback) -----------------
#if __has_builtin(__builtin_amdgcn_global_load_async_to_lds_b128)
#define HAVE_ASYNC 1
typedef __attribute__((address_space(1))) i32x4* gvec_p;   // global b128
typedef __attribute__((address_space(3))) i32x4* lvec_p;   // LDS b128
__device__ __forceinline__ void copy16_g2l(const void* g, void* l) {
  __builtin_amdgcn_global_load_async_to_lds_b128((gvec_p)g, (lvec_p)l, 0, 0);
}
#else
#define HAVE_ASYNC 0
__device__ __forceinline__ void copy16_g2l(const void* g, void* l) {
  *(uint4*)l = *(const uint4*)g;
}
#endif

#if __has_builtin(__builtin_amdgcn_s_wait_asynccnt)
#define WAIT_ASYNC(n) __builtin_amdgcn_s_wait_asynccnt(n)
#elif HAVE_ASYNC
#define WAIT_ASYNC(n) asm volatile("s_wait_asynccnt %0" ::"i"(n) : "memory")
#else
#define WAIT_ASYNC(n)
#endif

// ---- WMMA wrapper ----------------------------------------------------------
__device__ __forceinline__ v8f wmma_f16(v16h a, v16h b, v8f c) {
  return __builtin_amdgcn_wmma_f32_16x16x32_f16(false, a, false, b,
                                                (short)0, c, false, false);
}

// ---- Fragment loaders ------------------------------------------------------
// 16-bit A/B fragment for 16x16x32: lane L holds row/col (L&15), half h=L>>4.
// frag[0..7]  = mem[row*ld + h*8 .. +7], frag[8..15] = mem[row*ld + h*8+16 ..]
__device__ __forceinline__ v16h load_frag_h(const _Float16* base, int ld, int lane) {
  int r = lane & 15, h = lane >> 4;
  const v8h* p = (const v8h*)(base + (size_t)r * ld + h * 8);
  v8h lo = p[0];
  v8h hi = p[2];   // +16 halfs
  return __builtin_shufflevector(lo, hi, 0,1,2,3,4,5,6,7,8,9,10,11,12,13,14,15);
}

__device__ __forceinline__ v16h load_frag_f32(const float* base, int ld, int lane) {
  int r = lane & 15, h = lane >> 4;
  const float* q = base + (size_t)r * ld + h * 8;
  v16h f;
#pragma unroll
  for (int i = 0; i < 8; ++i) {
    f[i]     = (_Float16)q[i];
    f[8 + i] = (_Float16)q[16 + i];
  }
  return f;
}

// ---------------------------------------------------------------------------
// Kernel 1: QKV projection.  y = x @ W^T + b, f16 outputs (V transposed).
// W is converted f32->f16 once per block into LDS; B-fragments then come from
// ds_load_b128 instead of per-wave global f32 reads + conversions.
// ---------------------------------------------------------------------------
__global__ __launch_bounds__(256) void qkv_kernel(
    const float* __restrict__ x,
    const float* __restrict__ Wq, const float* __restrict__ bq,
    const float* __restrict__ Wk, const float* __restrict__ bk,
    const float* __restrict__ Wv, const float* __restrict__ bv,
    _Float16* __restrict__ Qh, _Float16* __restrict__ Kh,
    _Float16* __restrict__ Vt)
{
  __shared__ __align__(16) _Float16 wlds[DHEAD * DHEAD];   // 32KB, one W at a time

  int tid  = threadIdx.x;
  int lane = tid & 31;
  int wid  = blockIdx.x * 8 + (tid >> 5);   // 0..1023 row tiles
  int row0 = wid * 16;                       // flattened over B*N
  int b    = row0 / NSEQ;
  int n0   = row0 % NSEQ;
  int r = lane & 15, h = lane >> 4;

  // x tile 16x128 as four resident A-fragments
  v16h xa[4];
#pragma unroll
  for (int kb = 0; kb < 4; ++kb)
    xa[kb] = load_frag_f32(x + (size_t)row0 * DHEAD + kb * 32, DHEAD, lane);

  const float* Ws[3] = {Wq, Wk, Wv};
  const float* bs[3] = {bq, bk, bv};

#pragma unroll
  for (int m = 0; m < 3; ++m) {
    // ---- stage W[m] into LDS as f16 (block-cooperative) ----
    __syncthreads();   // previous m's fragment reads done
    for (int i = tid; i < DHEAD * DHEAD / 4; i += 256) {
      float4 f = ((const float4*)Ws[m])[i];
      _Float16* d = &wlds[i * 4];
      d[0] = (_Float16)f.x; d[1] = (_Float16)f.y;
      d[2] = (_Float16)f.z; d[3] = (_Float16)f.w;
    }
    __syncthreads();

#pragma unroll
    for (int et = 0; et < 8; ++et) {            // output column tiles of 16
      float bval = bs[m][et * 16 + r];          // C layout: N = lane&15
      v8f acc;
#pragma unroll
      for (int i = 0; i < 8; ++i) acc[i] = bval;
#pragma unroll
      for (int kb = 0; kb < 4; ++kb) {
        // B = W^T chunk: B[k][e] = W[e][k] -> row-major read of W rows.
        v16h wb = load_frag_h(&wlds[(et * 16) * DHEAD + kb * 32], DHEAD, lane);
        acc = wmma_f16(xa[kb], wb, acc);
      }
      // D layout: VGPR rr -> row (rr + 8*h), col (et*16 + lane&15)
#pragma unroll
      for (int rr = 0; rr < 8; ++rr) {
        int mrow = rr + 8 * h;
        int col  = et * 16 + r;
        _Float16 val = (_Float16)acc[rr];
        if (m == 0)
          Qh[(size_t)(row0 + mrow) * DHEAD + col] = val;
        else if (m == 1)
          Kh[(size_t)(row0 + mrow) * DHEAD + col] = val;
        else
          Vt[((size_t)b * DHEAD + col) * NSEQ + (n0 + mrow)] = val;
      }
    }
  }
}

// ---------------------------------------------------------------------------
// K/V chunk staging: K tile (KC x 128, contiguous 8KB) and V tile
// (128 x KC slice of transposed V) -> LDS.  4 x b128 per thread per chunk.
// ---------------------------------------------------------------------------
__device__ __forceinline__ void stage_chunk(const _Float16* __restrict__ Kb,
                                            const _Float16* __restrict__ Vb,
                                            int key0, _Float16* ks,
                                            _Float16* vs, int tid)
{
  // K: KC*DHEAD halfs = 512 uint4, contiguous in global.
  const char* gk = (const char*)(Kb + (size_t)key0 * DHEAD);
#pragma unroll
  for (int i = 0; i < 2; ++i) {
    int u = tid + i * 256;
    copy16_g2l(gk + u * 16, (char*)ks + u * 16);
  }
  // V: 128 rows (d) x KC keys, row stride NSEQ in global, KC in LDS.
#pragma unroll
  for (int i = 0; i < 2; ++i) {
    int u   = tid + i * 256;
    int row = u >> 2, seg = u & 3;            // 4 x 16B per row
    copy16_g2l(Vb + (size_t)row * NSEQ + key0 + seg * 8,
               vs + row * KC + seg * 8);
  }
}

// ---------------------------------------------------------------------------
// Kernel 2: flash attention.  8 waves/block share one batch; K/V chunks are
// staged block-wide in double-buffered LDS (async on CDNA5).  One wave owns
// 16 query rows with online softmax; P converts C-layout->A-fragment via a
// per-wave LDS tile.  Deterministic partial sums out.
// ---------------------------------------------------------------------------
__global__ __launch_bounds__(256) void attn_kernel(
    const _Float16* __restrict__ Qh, const _Float16* __restrict__ Kh,
    const _Float16* __restrict__ Vt, float* __restrict__ part)
{
  __shared__ __align__(16) _Float16 ksh[2][KC * DHEAD];   // 2 x 8KB
  __shared__ __align__(16) _Float16 vsh[2][DHEAD * KC];   // 2 x 8KB
  __shared__ __align__(16) _Float16 plds[8][16 * KC];     // 8KB

  int tid  = threadIdx.x;
  int lane = tid & 31;
  int w    = tid >> 5;
  int b    = blockIdx.x >> 5;                 // 32 blocks per batch
  int qt   = (blockIdx.x & 31) * 8 + w;       // 0..255
  int r = lane & 15, h = lane >> 4;

  const _Float16* Qb = Qh + ((size_t)b * NSEQ + (size_t)qt * 16) * DHEAD;
  const _Float16* Kb = Kh + (size_t)b * NSEQ * DHEAD;
  const _Float16* Vb = Vt + (size_t)b * DHEAD * NSEQ;

  // Resident Q fragments (16x128)
  v16h qa[4];
#pragma unroll
  for (int kb = 0; kb < 4; ++kb)
    qa[kb] = load_frag_h(Qb + kb * 32, DHEAD, lane);

  v8f   O[8];
  float rmax[8], rsum[8];
#pragma unroll
  for (int i = 0; i < 8; ++i) {
    O[i] = (v8f){0.f, 0.f, 0.f, 0.f, 0.f, 0.f, 0.f, 0.f};
    rmax[i] = -__builtin_inff();
    rsum[i] = 0.f;
  }

  // prologue: stage chunk 0 into buffer 0
  stage_chunk(Kb, Vb, 0, ksh[0], vsh[0], tid);

  for (int c = 0; c < NCHUNK; ++c) {
    int key0 = c * KC;
    int buf  = c & 1;

    __syncthreads();   // prior iteration done reading buf^1 (and plds)
    if (c + 1 < NCHUNK) {
      stage_chunk(Kb, Vb, key0 + KC, ksh[buf ^ 1], vsh[buf ^ 1], tid);
      WAIT_ASYNC(4);   // 4 newly issued outstanding -> buf's 4 have landed
    } else {
      WAIT_ASYNC(0);
    }
    __syncthreads();   // buf's data visible block-wide

    const _Float16* kt = ksh[buf];
    const _Float16* vt = vsh[buf];

    // ---- scores: S = Q * K^T, two 16-key tiles (from LDS) ----------------
    v8f S[2];
#pragma unroll
    for (int t = 0; t < 2; ++t) {
      v8f s = (v8f){0.f, 0.f, 0.f, 0.f, 0.f, 0.f, 0.f, 0.f};
#pragma unroll
      for (int kb = 0; kb < 4; ++kb) {
        v16h kf = load_frag_h(kt + (size_t)(t * 16) * DHEAD + kb * 32,
                              DHEAD, lane);
        s = wmma_f16(qa[kb], kf, s);
      }
      S[t] = s;
    }

    // ---- online softmax: row max / rescale (rows live in 16-lane halves) -
#pragma unroll
    for (int rr = 0; rr < 8; ++rr) {
      float tm = fmaxf(S[0][rr], S[1][rr]);
#pragma unroll
      for (int off = 1; off <= 8; off <<= 1)
        tm = fmaxf(tm, __shfl_xor(tm, off, 32));
      float nm = fmaxf(rmax[rr], tm);
      float sc = __expf(rmax[rr] - nm);
      rmax[rr] = nm;
      rsum[rr] *= sc;
#pragma unroll
      for (int dt = 0; dt < 8; ++dt) O[dt][rr] *= sc;
    }

    // ---- P = exp(S - max): C-layout -> A-fragment via per-wave LDS tile --
#pragma unroll
    for (int rr = 0; rr < 8; ++rr) {
      float p0 = __expf(S[0][rr] - rmax[rr]);
      float p1 = __expf(S[1][rr] - rmax[rr]);
      int mrow = rr + 8 * h;
      plds[w][mrow * KC + r]      = (_Float16)p0;
      plds[w][mrow * KC + 16 + r] = (_Float16)p1;
      float ps = p0 + p1;
#pragma unroll
      for (int off = 1; off <= 8; off <<= 1)
        ps += __shfl_xor(ps, off, 32);
      rsum[rr] += ps;
    }
    __syncthreads();
    v16h pf = load_frag_h(&plds[w][0], KC, lane);

    // ---- O += P * V (V from LDS: rows = d, ld = KC) ----------------------
#pragma unroll
    for (int dt = 0; dt < 8; ++dt) {
      v16h vf = load_frag_h(vt + (size_t)(dt * 16) * KC, KC, lane);
      O[dt] = wmma_f16(pf, vf, O[dt]);
    }
  }

  // ---- finalize: normalize rows, sum the 16 query rows ---------------------
#pragma unroll
  for (int dt = 0; dt < 8; ++dt) {
    float cs = 0.f;
#pragma unroll
    for (int rr = 0; rr < 8; ++rr) cs += O[dt][rr] / rsum[rr];
    cs += __shfl_xor(cs, 16, 32);
    if (lane < 16)
      part[((size_t)(b * QTPB + qt)) * DHEAD + dt * 16 + lane] = cs;
  }
}

// ---------------------------------------------------------------------------
// Kernel 3: deterministic mean over query tiles.
// ---------------------------------------------------------------------------
__global__ void reduce_kernel(const float* __restrict__ part,
                              float* __restrict__ out) {
  int b = blockIdx.x, d = threadIdx.x;
  float s = 0.f;
  for (int qt = 0; qt < QTPB; ++qt)
    s += part[((size_t)(b * QTPB + qt)) * DHEAD + d];
  out[b * DHEAD + d] = s * (1.0f / NSEQ);
}

// ---------------------------------------------------------------------------
extern "C" void kernel_launch(void* const* d_in, const int* in_sizes, int n_in,
                              void* d_out, int out_size, void* d_ws, size_t ws_size,
                              hipStream_t stream) {
  const float* x  = (const float*)d_in[0];
  const float* Wq = (const float*)d_in[1];
  const float* bq = (const float*)d_in[2];
  const float* Wk = (const float*)d_in[3];
  const float* bk = (const float*)d_in[4];
  const float* Wv = (const float*)d_in[5];
  const float* bv = (const float*)d_in[6];
  float* out = (float*)d_out;

  // Workspace: Qh(4MB) | Kh(4MB) | Vt(4MB, transposed) | partials(512KB)
  const size_t QKV_BYTES = (size_t)NB * NSEQ * DHEAD * sizeof(_Float16);
  char* ws = (char*)d_ws;
  _Float16* Qh  = (_Float16*)(ws);
  _Float16* Kh  = (_Float16*)(ws + QKV_BYTES);
  _Float16* Vt  = (_Float16*)(ws + 2 * QKV_BYTES);
  float*    prt = (float*)   (ws + 3 * QKV_BYTES);

  qkv_kernel<<<128, 256, 0, stream>>>(x, Wq, bq, Wk, bk, Wv, bv, Qh, Kh, Vt);
  attn_kernel<<<128, 256, 0, stream>>>(Qh, Kh, Vt, prt);
  reduce_kernel<<<NB, DHEAD, 0, stream>>>(prt, out);
}
